// RNNGAN_34376918238006
// MI455X (gfx1250) — matrix operand order repaired and checked by reading, not compiled, not run on hardware
//
#include <hip/hip_runtime.h>
#include <hip/hip_bf16.h>

// ---------------------------------------------------------------------------
// Problem constants
// ---------------------------------------------------------------------------
#define NN   256          // batch elements
#define TT   32           // timesteps
#define FF   512          // input features
#define HH   512          // hidden
#define H3   1536         // 3*H
#define NT   (NN * TT)    // 8192 rows for the hoisted GEMM

typedef __attribute__((ext_vector_type(8)))  __bf16 v8bf;
typedef __attribute__((ext_vector_type(16))) __bf16 v16bf;
typedef __attribute__((ext_vector_type(8)))  float  v8f;

// ---------------------------------------------------------------------------
// Helpers
// ---------------------------------------------------------------------------
__device__ __forceinline__ __bf16 to_bf16(float f) {
    union { float f; unsigned u; } v; v.f = f;
    unsigned r = v.u + (0x7FFFu + ((v.u >> 16) & 1u));   // RNE
    union { unsigned short s; __bf16 b; } o;
    o.s = (unsigned short)(r >> 16);
    return o.b;
}

__device__ __forceinline__ v8f wmma_bf16(v16bf a, v16bf b, v8f c) {
    // v_wmma_f32_16x16x32_bf16  (D = A x B + C, f32 accumulate)
    return __builtin_amdgcn_wmma_f32_16x16x32_bf16(
        false, a, false, b, (short)0, c, false, false);
}

// A-fragment (16x32 bf16, M x K): lane&15 = row; low half-wave holds K chunks
// {k0..k0+7, k0+16..k0+23}, high half-wave holds {k0+8..15, k0+24..31}.
__device__ __forceinline__ v16bf frag_a(const __bf16* __restrict__ row, int k0, bool hi) {
    const __bf16* p = row + k0 + (hi ? 8 : 0);
    v8bf lo = *(const v8bf*)p;
    v8bf hh = *(const v8bf*)(p + 16);
    v16bf f;
#pragma unroll
    for (int i = 0; i < 8; ++i) { f[i] = lo[i]; f[i + 8] = hh[i]; }
    return f;
}

// B-fragment (32x16 bf16, K x N): lane&15 = column; low half-wave holds
// K chunks {k0..k0+7, k0+16..k0+23}, high half-wave {k0+8..15, k0+24..31}.
__device__ __forceinline__ v16bf frag_b(const __bf16* __restrict__ row, int k0, bool hi) {
    const __bf16* p = row + k0 + (hi ? 8 : 0);
    v8bf lo = *(const v8bf*)p;
    v8bf hh = *(const v8bf*)(p + 16);
    v16bf f;
#pragma unroll
    for (int i = 0; i < 8; ++i) { f[i] = lo[i]; f[i + 8] = hh[i]; }
    return f;
}

__device__ __forceinline__ float sigmoidf_(float x) {
    return 1.0f / (1.0f + __expf(-x));
}

// ---------------------------------------------------------------------------
// Kernel 0: init (dates -> out as float, zero global-barrier state)
// ---------------------------------------------------------------------------
__global__ void k_init(const int* __restrict__ dates, float* __restrict__ out,
                       unsigned* __restrict__ bar) {
    int i = blockIdx.x * blockDim.x + threadIdx.x;
    if (i < NN) out[i] = (float)dates[i];
    if (i < 4)  bar[i] = 0u;
}

// ---------------------------------------------------------------------------
// Kernel 1: f32 -> bf16 conversion (x and the 4 weight matrices)
// ---------------------------------------------------------------------------
__global__ void k_cvt(const float* __restrict__ src, __bf16* __restrict__ dst, int n) {
    int i      = blockIdx.x * blockDim.x + threadIdx.x;
    int stride = gridDim.x * blockDim.x;
    for (; i < n; i += stride) dst[i] = to_bf16(src[i]);
}

// ---------------------------------------------------------------------------
// Kernel 2: dense GEMM  GI0 = X(8192x512) @ Wih0^T(512->1536) + bih0  (WMMA)
// Block = 256 threads = 8 waves arranged 4(M) x 2(N); WG tile 128x128.
// Each wave: 32(M) x 64(N) = 2x4 accumulators, K-loop of 16 WMMAs each.
// ---------------------------------------------------------------------------
__global__ __launch_bounds__(256) void k_gemm_gi0(
    const __bf16* __restrict__ xb,   // 8192 x 512, row-major
    const __bf16* __restrict__ w,    // 1536 x 512, row-major (Wih0)
    const float*  __restrict__ bias, // 1536
    float*        __restrict__ gi0)  // 8192 x 1536
{
    const int lane = threadIdx.x & 31;
    const int wid  = threadIdx.x >> 5;
    const int wm   = wid & 3;
    const int wn   = wid >> 2;
    const int m0   = blockIdx.y * 128 + wm * 32;
    const int n0   = blockIdx.x * 128 + wn * 64;
    const bool hi  = lane >= 16;
    const int l15  = lane & 15;

    v8f c[2][4];
#pragma unroll
    for (int i = 0; i < 2; ++i)
#pragma unroll
        for (int j = 0; j < 4; ++j)
            c[i][j] = (v8f){0.f, 0.f, 0.f, 0.f, 0.f, 0.f, 0.f, 0.f};

    const __bf16* arow0 = xb + (size_t)(m0 + l15) * FF;
    const __bf16* arow1 = arow0 + (size_t)16 * FF;

#pragma unroll 4
    for (int k0 = 0; k0 < FF; k0 += 32) {
        v16bf a[2], b[4];
        a[0] = frag_a(arow0, k0, hi);
        a[1] = frag_a(arow1, k0, hi);
#pragma unroll
        for (int j = 0; j < 4; ++j)
            b[j] = frag_b(w + (size_t)(n0 + j * 16 + l15) * FF, k0, hi);
#pragma unroll
        for (int i = 0; i < 2; ++i)
#pragma unroll
            for (int j = 0; j < 4; ++j)
                c[i][j] = wmma_bf16(a[i], b[j], c[i][j]);
    }

    // C/D layout: VGPR r -> M = r (lanes 0-15) / 8+r (lanes 16-31); N = lane&15
#pragma unroll
    for (int j = 0; j < 4; ++j) {
        const int colg = n0 + j * 16 + l15;
        const float bv = bias[colg];
#pragma unroll
        for (int i = 0; i < 2; ++i) {
#pragma unroll
            for (int rr = 0; rr < 8; ++rr) {
                const int row = m0 + i * 16 + rr + (hi ? 8 : 0);
                gi0[(size_t)row * H3 + colg] = c[i][j][rr] + bv;
            }
        }
    }
}

// ---------------------------------------------------------------------------
// Recurrence kernel: 8 persistent workgroups x 512 threads (16 waves each).
// h vectors live per-WG in LDS; gate pre-activations (gh0/gh1/gi1, 1536 f32
// each) go through global memory; 3 device-scope barriers per step.
// ---------------------------------------------------------------------------
#define RWG     8
#define RTHR    512
#define RWAVES  (RWG * (RTHR / 32))   // 128 wave jobs

__device__ __forceinline__ void gbarrier(unsigned* cnt, unsigned* gen) {
    __threadfence();
    __syncthreads();
    if (threadIdx.x == 0) {
        volatile unsigned* vgen = gen;
        unsigned g = *vgen;                 // read generation BEFORE arriving
        if (atomicAdd(cnt, 1u) == (unsigned)(RWG - 1)) {
            atomicExch(cnt, 0u);
            __threadfence();
            atomicAdd(gen, 1u);
        } else {
            while (*vgen == g) { __builtin_amdgcn_s_sleep(2); }
        }
    }
    __syncthreads();
    __threadfence();
}

// One 16-column slice of a 1x512 @ 512x1536 matvec via M-padded WMMA.
// A has only row 0 nonzero -> only lanes 0 and 16 carry data.
__device__ __forceinline__ void matvec_tile(
    const __bf16* __restrict__ W,   // 1536 x 512 row-major
    const __bf16* __restrict__ hv,  // 512 bf16 (LDS)
    float* __restrict__ dst, int n0, int lane)
{
    const bool hi = lane >= 16;
    const int  l15 = lane & 15;
    const __bf16* wrow = W + (size_t)(n0 + l15) * HH;
    v8f c = (v8f){0.f, 0.f, 0.f, 0.f, 0.f, 0.f, 0.f, 0.f};
#pragma unroll 4
    for (int k0 = 0; k0 < HH; k0 += 32) {
        v16bf b = frag_b(wrow, k0, hi);
        v16bf a;
        if (l15 == 0) {
            a = frag_a(hv, k0, hi);
        } else {
#pragma unroll
            for (int i = 0; i < 16; ++i) a[i] = static_cast<__bf16>(0.0f);
        }
        c = wmma_bf16(a, b, c);   // EXEC all-ones here (uniform control flow)
    }
    if (!hi) dst[n0 + l15] = c[0];   // D row M=0 lives in VGPR0, lanes 0-15
}

__global__ __launch_bounds__(RTHR) void k_recurrence(
    const int*    __restrict__ dates,
    const float*  __restrict__ gi0,    // 8192 x 1536 (bih0 already added)
    const __bf16* __restrict__ Whh0,
    const __bf16* __restrict__ Wih1,
    const __bf16* __restrict__ Whh1,
    const float*  __restrict__ bih1,
    const float*  __restrict__ bhh0,
    const float*  __restrict__ bhh1,
    float*        __restrict__ out,    // [256 dates][256x512 states]
    float*        __restrict__ gh0g,   // 1536
    float*        __restrict__ gh1g,   // 1536
    float*        __restrict__ gi1g,   // 1536
    unsigned*     __restrict__ bar)
{
    __shared__ __attribute__((aligned(16))) __bf16 h0b[HH];
    __shared__ __attribute__((aligned(16))) __bf16 h1b[HH];
    __shared__ float h0f[HH];
    __shared__ float h1f[HH];

    const int tid  = threadIdx.x;
    const int lane = tid & 31;
    const int wid  = tid >> 5;
    const int gw   = blockIdx.x * (RTHR / 32) + wid;   // 0..127 global wave id

    if (tid < HH) {
        h0f[tid] = 0.0f; h1f[tid] = 0.0f;
        h0b[tid] = static_cast<__bf16>(0.0f);
        h1b[tid] = static_cast<__bf16>(0.0f);
    }
    __syncthreads();

    for (int n = 0; n < NN; ++n) {
        const int  d    = dates[n];
        const bool same = (n > 0) && (d == dates[n - 1]);

        if (!same) {
            for (int t = 0; t < TT; ++t) {
                const float* gi = gi0 + (size_t)(n * TT + t) * H3;

                // Phase A: gh0 = h0 @ Whh0^T, gh1 = h1 @ Whh1^T (192 tile jobs)
                for (int j = gw; j < 192; j += RWAVES) {
                    if (j < 96) matvec_tile(Whh0, h0b, gh0g, j * 16, lane);
                    else        matvec_tile(Whh1, h1b, gh1g, (j - 96) * 16, lane);
                }
                gbarrier(bar, bar + 1);

                // Phase B (redundant per WG): layer-0 gates -> h0_new
                if (tid < HH) {
                    const float ir = gi[tid];
                    const float iz = gi[HH + tid];
                    const float in_ = gi[2 * HH + tid];
                    const float hr = gh0g[tid]          + bhh0[tid];
                    const float hz = gh0g[HH + tid]     + bhh0[HH + tid];
                    const float hn = gh0g[2 * HH + tid] + bhh0[2 * HH + tid];
                    const float r  = sigmoidf_(ir + hr);
                    const float z  = sigmoidf_(iz + hz);
                    const float nn = tanhf(in_ + r * hn);
                    const float h  = (1.0f - z) * nn + z * h0f[tid];
                    h0f[tid] = h;
                    h0b[tid] = to_bf16(h);
                }
                __syncthreads();

                // Phase C: gi1 = h0_new @ Wih1^T (96 tile jobs)
                for (int j = gw; j < 96; j += RWAVES)
                    matvec_tile(Wih1, h0b, gi1g, j * 16, lane);
                gbarrier(bar, bar + 1);

                // Phase D (redundant per WG): layer-1 gates -> h1_new
                if (tid < HH) {
                    const float ir = gi1g[tid]          + bih1[tid];
                    const float iz = gi1g[HH + tid]     + bih1[HH + tid];
                    const float in_ = gi1g[2 * HH + tid] + bih1[2 * HH + tid];
                    const float hr = gh1g[tid]          + bhh1[tid];
                    const float hz = gh1g[HH + tid]     + bhh1[HH + tid];
                    const float hn = gh1g[2 * HH + tid] + bhh1[2 * HH + tid];
                    const float r  = sigmoidf_(ir + hr);
                    const float z  = sigmoidf_(iz + hz);
                    const float nn = tanhf(in_ + r * hn);
                    const float h  = (1.0f - z) * nn + z * h1f[tid];
                    h1f[tid] = h;
                    h1b[tid] = to_bf16(h);
                }
                __syncthreads();
                gbarrier(bar, bar + 1);   // protect gh*/gi* vs next step's writes
            }
        }

        // states[n] = last-layer hidden (identical in every WG; WG0 writes)
        if (blockIdx.x == 0 && tid < HH)
            out[NN + (size_t)n * HH + tid] = h1f[tid];
    }
}

// ---------------------------------------------------------------------------
// Host-side launch
// ---------------------------------------------------------------------------
extern "C" void kernel_launch(void* const* d_in, const int* in_sizes, int n_in,
                              void* d_out, int out_size, void* d_ws, size_t ws_size,
                              hipStream_t stream) {
    const int*   dates = (const int*)  d_in[0];
    const float* x     = (const float*)d_in[1];
    const float* Wih0  = (const float*)d_in[2];
    const float* Whh0  = (const float*)d_in[3];
    const float* bih0  = (const float*)d_in[4];
    const float* bhh0  = (const float*)d_in[5];
    const float* Wih1  = (const float*)d_in[6];
    const float* Whh1  = (const float*)d_in[7];
    const float* bih1  = (const float*)d_in[8];
    const float* bhh1  = (const float*)d_in[9];
    float* out = (float*)d_out;

    // Workspace carve-up (256B aligned)
    char*  w   = (char*)d_ws;
    size_t off = 0;
    auto alloc = [&](size_t bytes) -> void* {
        void* p = w + off;
        off = (off + bytes + 255) & ~(size_t)255;
        return p;
    };
    __bf16* xb     = (__bf16*)alloc((size_t)NT * FF * 2);
    __bf16* Wih0b  = (__bf16*)alloc((size_t)H3 * FF * 2);
    __bf16* Whh0b  = (__bf16*)alloc((size_t)H3 * HH * 2);
    __bf16* Wih1b  = (__bf16*)alloc((size_t)H3 * HH * 2);
    __bf16* Whh1b  = (__bf16*)alloc((size_t)H3 * HH * 2);
    float*  gi0    = (float*) alloc((size_t)NT * H3 * 4);
    float*  gh0g   = (float*) alloc((size_t)H3 * 4);
    float*  gh1g   = (float*) alloc((size_t)H3 * 4);
    float*  gi1g   = (float*) alloc((size_t)H3 * 4);
    unsigned* bar  = (unsigned*)alloc(4 * sizeof(unsigned));

    // 0) dates -> out, reset barrier state (must be re-zeroed every launch)
    k_init<<<1, 256, 0, stream>>>(dates, out, bar);

    // 1) f32 -> bf16 conversions
    k_cvt<<<1024, 256, 0, stream>>>(x,    xb,    NT * FF);
    k_cvt<<<512,  256, 0, stream>>>(Wih0, Wih0b, H3 * FF);
    k_cvt<<<512,  256, 0, stream>>>(Whh0, Whh0b, H3 * HH);
    k_cvt<<<512,  256, 0, stream>>>(Wih1, Wih1b, H3 * HH);
    k_cvt<<<512,  256, 0, stream>>>(Whh1, Whh1b, H3 * HH);

    // 2) hoisted dense GEMM: GI0 = X @ Wih0^T + bih0  (WMMA bf16, f32 acc)
    k_gemm_gi0<<<dim3(H3 / 128, NT / 128), 256, 0, stream>>>(xb, Wih0b, bih0, gi0);

    // 3) sequential GRU chain, 8 persistent WGs cooperating via global barrier
    k_recurrence<<<RWG, RTHR, 0, stream>>>(dates, gi0, Whh0b, Wih1b, Whh1b,
                                           bih1, bhh0, bhh1, out,
                                           gh0g, gh1g, gi1g, bar);
    (void)in_sizes; (void)n_in; (void)out_size; (void)ws_size;
}
